// VectorQuantizer_48704929137135
// MI455X (gfx1250) — compile-verified, matrix operand order checked by reading, not run on hardware
//
#include <hip/hip_runtime.h>
#include <hip/hip_bf16.h>

// ---- problem constants (fixed by setup_inputs) ----
constexpr int BB = 16, LL = 2048, DD = 256, KK = 4096;
constexpr int NN = BB * LL;                  // 32768 rows
constexpr int ROWS = 64;                     // rows per workgroup (4 subtiles of 16)
constexpr int CT = 128;                      // codes per LDS tile (8 subtiles of 16)
constexpr int TILES = KK / CT;               // 32
constexpr int KSTEPS = DD / 32;              // 8 WMMA k-steps
constexpr int CSUB_PER_WAVE = 4;             // code subtiles per wave
constexpr float TEMP_INV = 1.0f;             // TEMP = 1
constexpr float EPS = 1e-10f;

// d_out layout (flat concat of reference outputs, all f32)
constexpr size_t ZQ_OFF = 0;
constexpr size_t IDX_OFF = (size_t)NN * DD;            // 8388608
constexpr size_t COMMIT_OFF = IDX_OFF + NN;            // 8421376
constexpr size_t ENT_OFF = COMMIT_OFF + 1;             // 8421377

// workspace layout (bytes)
constexpr size_t WS_CBH = 0;                                   // bf16 codebook: K*D*2 = 2 MB
constexpr size_t WS_CNORM = (size_t)KK * DD * 2;               // f32 [K]
constexpr size_t WS_GAVG = WS_CNORM + (size_t)KK * 4;          // f32 [K]
constexpr size_t WS_GCOMMIT = WS_GAVG + (size_t)KK * 4;        // f32 [1]

typedef __attribute__((ext_vector_type(16))) __bf16 v16bf;
typedef __attribute__((ext_vector_type(8)))  float  v8f;

// ---- dynamic-LDS carve offsets (bytes) ----
constexpr int SM_ZHI = 0;                      // [ROWS][D] bf16 = 32 KB
constexpr int SM_ZLO = SM_ZHI + ROWS * DD * 2; // 32 KB
constexpr int SM_CSH = SM_ZLO + ROWS * DD * 2; // [CT][D] bf16 = 64 KB
constexpr int SM_AVG = SM_CSH + CT * DD * 2;   // f32 [K] = 16 KB
constexpr int SM_RMIN = SM_AVG + KK * 4;       // f32 [ROWS]
constexpr int SM_RS = SM_RMIN + ROWS * 4;      // f32 [ROWS]
constexpr int SM_RIDX = SM_RS + ROWS * 4;      // i32 [ROWS]
constexpr int SMEM_BYTES = SM_RIDX + ROWS * 4; // 148224 B (<= 320 KB/WG on CDNA5)

__device__ inline v16bf ld_frag(const __bf16* p) {
    union { v16bf v; uint4 q[2]; } u;
    u.q[0] = *reinterpret_cast<const uint4*>(p);        // K chunk 0 (8 bf16)
    u.q[1] = *reinterpret_cast<const uint4*>(p + 16);   // K chunk 1 (8 bf16)
    return u.v;
}

// Wave macro-tile: 16 rows x 64 codes (4 accumulators), bf16 hi+lo split GEMM.
// Per k-step: 4 A-frag loads + 8 B-frag loads feed 8 WMMAs (1.5 ds_load/WMMA).
__device__ inline void gemm_tile4(const __bf16* __restrict__ zhi,
                                  const __bf16* __restrict__ zlo,
                                  const __bf16* __restrict__ csh,
                                  int arow_off, int brow_off0,
                                  v8f acc[CSUB_PER_WAVE]) {
#pragma unroll
    for (int c = 0; c < CSUB_PER_WAVE; ++c)
        acc[c] = (v8f){0.f, 0.f, 0.f, 0.f, 0.f, 0.f, 0.f, 0.f};
#pragma unroll
    for (int ks = 0; ks < KSTEPS; ++ks) {
        const int o = ks * 32;
        v16bf ah = ld_frag(zhi + arow_off + o);
        v16bf al = ld_frag(zlo + arow_off + o);
#pragma unroll
        for (int c = 0; c < CSUB_PER_WAVE; ++c) {
            v16bf b = ld_frag(csh + brow_off0 + c * 16 * DD + o);
            acc[c] = __builtin_amdgcn_wmma_f32_16x16x32_bf16(false, ah, false, b,
                                                             (short)0, acc[c], false, false);
            acc[c] = __builtin_amdgcn_wmma_f32_16x16x32_bf16(false, al, false, b,
                                                             (short)0, acc[c], false, false);
        }
    }
}

// ---- prep: bf16 codebook + exact f32 ||c||^2 ; also zero accumulators ----
__global__ __launch_bounds__(256) void vq_prep(const float* __restrict__ cb,
                                               __bf16* __restrict__ cbh,
                                               float* __restrict__ cnorm,
                                               float* __restrict__ gAvg,
                                               float* __restrict__ gCommit) {
    __shared__ float red[256];
    const int k = blockIdx.x;          // 4096 blocks
    const int t = threadIdx.x;         // 256 = D
    const float x = cb[(size_t)k * DD + t];
    cbh[(size_t)k * DD + t] = (__bf16)x;
    red[t] = x * x;
    __syncthreads();
    for (int s = 128; s > 0; s >>= 1) {
        if (t < s) red[t] += red[t + s];
        __syncthreads();
    }
    if (t == 0) {
        cnorm[k] = red[0];
        gAvg[k] = 0.f;
        if (k == 0) gCommit[0] = 0.f;
    }
}

// ---- main fused kernel: GEMM + online softmax + argmin + gather + commit ----
__global__ __launch_bounds__(256) void vq_main(const float* __restrict__ z,
                                               const float* __restrict__ cb,
                                               const __bf16* __restrict__ cbh,
                                               const float* __restrict__ cnorm,
                                               float* __restrict__ out,
                                               float* __restrict__ gAvg,
                                               float* __restrict__ gCommit) {
    extern __shared__ char smem[];
    __bf16* zhi = (__bf16*)(smem + SM_ZHI);
    __bf16* zlo = (__bf16*)(smem + SM_ZLO);
    __bf16* csh = (__bf16*)(smem + SM_CSH);
    float*  avg = (float*)(smem + SM_AVG);
    float*  rmin = (float*)(smem + SM_RMIN);
    float*  rS   = (float*)(smem + SM_RS);
    int*    ridx = (int*)(smem + SM_RIDX);
    // lane-reduction scratch + commit scratch alias the code tile (idle between passes)
    float* redm = (float*)csh;                  // [ROWS][32]
    float* redS = redm + ROWS * 32;             // [ROWS][32]
    int*   redi = (int*)(redS + ROWS * 32);     // [ROWS][32]
    float* csc  = (float*)(redi + ROWS * 32);   // [256]

    const int tid = threadIdx.x;
    const int lane = tid & 31;
    const int wave = tid >> 5;
    const int row_sub = wave & 3;          // 4 row subtiles of 16
    const int cgrp = wave >> 2;            // 2 code groups of 4 subtiles (64 codes)
    const int wg_row0 = blockIdx.x * ROWS;

    // fragment base offsets (elements) into LDS tiles
    const int arow_off  = (row_sub * 16 + (lane & 15)) * DD + (lane >> 4) * 8;
    const int brow_off0 = (cgrp * 64 + (lane & 15)) * DD + (lane >> 4) * 8;

    // load z tile, split into bf16 hi/lo; zero LDS avg accumulator
    {
        const float4* zsrc = (const float4*)(z + (size_t)wg_row0 * DD);
        for (int i = tid; i < ROWS * DD / 4; i += 256) {
            float4 v = zsrc[i];
            const int e = i * 4;
            float xs[4] = {v.x, v.y, v.z, v.w};
#pragma unroll
            for (int c = 0; c < 4; ++c) {
                __bf16 h = (__bf16)xs[c];
                zhi[e + c] = h;
                zlo[e + c] = (__bf16)(xs[c] - (float)h);
            }
        }
        for (int i = tid; i < KK; i += 256) avg[i] = 0.f;
    }

    // per-lane online state (8 rows per lane in each C tile)
    float mn[8], sm[8];
    int bi[8];
#pragma unroll
    for (int j = 0; j < 8; ++j) { mn[j] = 1e30f; sm[j] = 0.f; bi[j] = 0; }

    v8f acc[CSUB_PER_WAVE];

    // ---------------- pass 1: argmin + online softmax (m, S) ----------------
    for (int t = 0; t < TILES; ++t) {
        __syncthreads();
        {
            const uint4* src = (const uint4*)(cbh + (size_t)t * CT * DD);
            uint4* dst = (uint4*)csh;
            for (int i = tid; i < CT * DD / 8; i += 256) dst[i] = src[i];
            if (t + 1 < TILES)
                __builtin_prefetch((const char*)(cbh + (size_t)(t + 1) * CT * DD) + tid * 256, 0, 0);
        }
        __syncthreads();
        gemm_tile4(zhi, zlo, csh, arow_off, brow_off0, acc);
#pragma unroll
        for (int c = 0; c < CSUB_PER_WAVE; ++c) {
            const int k = t * CT + cgrp * 64 + c * 16 + (lane & 15);
            const float cn = cnorm[k];
#pragma unroll
            for (int j = 0; j < 8; ++j) {
                const float d = cn - 2.0f * acc[c][j];   // dist minus row-const ||z||^2
                if (d < mn[j]) {
                    sm[j] = sm[j] * __expf((d - mn[j]) * TEMP_INV) + 1.0f;
                    mn[j] = d;
                    bi[j] = k;
                } else {
                    sm[j] += __expf((mn[j] - d) * TEMP_INV);
                }
            }
        }
    }

    // cross-lane / cross-wave combine (per row: 2 cgrps x 16 cols = 32 partials)
    __syncthreads();
    {
        const int slot = cgrp * 16 + (lane & 15);
        const int rbase = row_sub * 16 + (lane >> 4) * 8;
#pragma unroll
        for (int j = 0; j < 8; ++j) {
            const int r = rbase + j;
            redm[r * 32 + slot] = mn[j];
            redS[r * 32 + slot] = sm[j];
            redi[r * 32 + slot] = bi[j];
        }
    }
    __syncthreads();
    if (tid < ROWS) {
        float m = 1e30f; int ix = 0;
        for (int s = 0; s < 32; ++s) {
            const float v = redm[tid * 32 + s];
            if (v < m) { m = v; ix = redi[tid * 32 + s]; }
        }
        float S = 0.f;
        for (int s = 0; s < 32; ++s)
            S += redS[tid * 32 + s] * __expf((m - redm[tid * 32 + s]) * TEMP_INV);
        rmin[tid] = m; rS[tid] = S; ridx[tid] = ix;
    }
    __syncthreads();

    // ---------------- gather z_q, commit loss, indices ----------------
    {
        const int r = tid >> 2;            // 4 threads per row
        const int part = tid & 3;          // 64 d-elems each (16 x float4)
        const int ix = ridx[r];
        const int grow = wg_row0 + r;
        const float4* crow = (const float4*)(cb + (size_t)ix * DD);
        const float4* zrow = (const float4*)(z + (size_t)grow * DD);
        float4* orow = (float4*)(out + ZQ_OFF + (size_t)grow * DD);
        float cacc = 0.f;
#pragma unroll
        for (int q = part * 16; q < part * 16 + 16; ++q) {
            float4 c4 = crow[q], z4 = zrow[q];
            orow[q] = c4;                   // z_q_st == z_q numerically
            float dx = z4.x - c4.x, dy = z4.y - c4.y, dz = z4.z - c4.z, dw = z4.w - c4.w;
            cacc += dx * dx + dy * dy + dz * dz + dw * dw;
        }
        if (part == 0) out[IDX_OFF + grow] = (float)ix;
        csc[tid] = cacc;
        __syncthreads();
        for (int s = 128; s > 0; s >>= 1) {
            if (tid < s) csc[tid] += csc[tid + s];
            __syncthreads();
        }
        if (tid == 0) atomicAdd(gCommit, csc[0]);
    }

    // per-lane row constants for pass 2
    float lrm[8], linv[8];
    {
        const int rbase = row_sub * 16 + (lane >> 4) * 8;
#pragma unroll
        for (int j = 0; j < 8; ++j) {
            lrm[j] = rmin[rbase + j];
            linv[j] = 1.0f / rS[rbase + j];
        }
    }

    // ---------------- pass 2: soft-assign accumulation ----------------
    for (int t = 0; t < TILES; ++t) {
        __syncthreads();
        {
            const uint4* src = (const uint4*)(cbh + (size_t)t * CT * DD);
            uint4* dst = (uint4*)csh;
            for (int i = tid; i < CT * DD / 8; i += 256) dst[i] = src[i];
            if (t + 1 < TILES)
                __builtin_prefetch((const char*)(cbh + (size_t)(t + 1) * CT * DD) + tid * 256, 0, 0);
        }
        __syncthreads();
        gemm_tile4(zhi, zlo, csh, arow_off, brow_off0, acc);
#pragma unroll
        for (int c = 0; c < CSUB_PER_WAVE; ++c) {
            const int k = t * CT + cgrp * 64 + c * 16 + (lane & 15);
            const float cn = cnorm[k];
            float psum = 0.f;
#pragma unroll
            for (int j = 0; j < 8; ++j) {
                const float d = cn - 2.0f * acc[c][j];
                psum += __expf((lrm[j] - d) * TEMP_INV) * linv[j];
            }
            atomicAdd(&avg[k], psum);      // LDS ds_add_f32
        }
    }

    __syncthreads();
    for (int i = tid; i < KK; i += 256)
        if (avg[i] != 0.f) atomicAdd(&gAvg[i], avg[i]);
}

// ---- final: entropy loss over avg_assign + commit scale ----
__global__ __launch_bounds__(256) void vq_final(const float* __restrict__ gAvg,
                                                const float* __restrict__ gCommit,
                                                float* __restrict__ out) {
    __shared__ float red[256];
    const int t = threadIdx.x;
    float s = 0.f;
    for (int i = t; i < KK; i += 256) {
        const float a = gAvg[i] * (1.0f / (float)NN);   // mean over N
        s += a * __logf(a + EPS);
    }
    red[t] = s;
    __syncthreads();
    for (int d = 128; d > 0; d >>= 1) {
        if (t < d) red[t] += red[t + d];
        __syncthreads();
    }
    if (t == 0) {
        out[ENT_OFF] = red[0];
        out[COMMIT_OFF] = gCommit[0] * (1.0f / ((float)NN * (float)DD));
    }
}

extern "C" void kernel_launch(void* const* d_in, const int* in_sizes, int n_in,
                              void* d_out, int out_size, void* d_ws, size_t ws_size,
                              hipStream_t stream) {
    (void)in_sizes; (void)n_in; (void)out_size; (void)ws_size;
    const float* z  = (const float*)d_in[0];   // [16,2048,256] f32
    const float* cb = (const float*)d_in[1];   // [4096,256]   f32
    float* out = (float*)d_out;

    char* ws = (char*)d_ws;
    __bf16* cbh   = (__bf16*)(ws + WS_CBH);
    float* cnorm  = (float*)(ws + WS_CNORM);
    float* gAvg   = (float*)(ws + WS_GAVG);
    float* gCommit= (float*)(ws + WS_GCOMMIT);

    vq_prep<<<KK, 256, 0, stream>>>(cb, cbh, cnorm, gAvg, gCommit);
    vq_main<<<NN / ROWS, 256, SMEM_BYTES, stream>>>(z, cb, cbh, cnorm, out, gAvg, gCommit);
    vq_final<<<1, 256, 0, stream>>>(gAvg, gCommit, out);
}
